// BlockGlobalAttentionProduct_72121090834680
// MI455X (gfx1250) — matrix operand order repaired and checked by reading, not compiled, not run on hardware
//
#include <hip/hip_runtime.h>

typedef __attribute__((ext_vector_type(16))) _Float16 v16h;
typedef __attribute__((ext_vector_type(8)))  _Float16 v8h;
typedef __attribute__((ext_vector_type(8)))  float    v8f;
typedef __attribute__((ext_vector_type(4)))  float    v4f;

// global-address-space pointer types: force global_load_* (LOADcnt only) instead of
// flat_load_* (LOADcnt+DScnt) for pointers reconstructed from the LDS pointer table.
typedef const float __attribute__((address_space(1))) gcf;
typedef const v4f   __attribute__((address_space(1))) gcv4f;

#define NGLOBC  64
#define DHEAD   64
#define LKEYS   192
#define GKEYS   384
#define TOTKEYS 640
#define KT      32
#define NIT     (TOTKEYS / KT)   // 20

// LDS strides in halves (all byte offsets 16B-aligned)
#define QS_STR 72
#define KS_STR 72
#define VT_STR 40
#define PS_STR 40

__device__ __forceinline__ v16h cat8(v8h lo, v8h hi) {
  return __builtin_shufflevector(lo, hi, 0,1,2,3,4,5,6,7,8,9,10,11,12,13,14,15);
}

// A-operand (16x32 f16, M x K). lane(0-15): m=lane, kh=0; lane(16-31): m=lane-16, kh=1.
// dwords 0-3 hold K = k0 + 8*kh + {0..7}; dwords 4-7 hold K = k0 + 16 + 8*kh + {0..7}.
__device__ __forceinline__ v16h loadA(const _Float16* buf, int stride, int m, int k0, int lane) {
  const int kh = lane >> 4;
  const _Float16* p = buf + m * stride + k0 + 8 * kh;
  v8h lo = *(const v8h*)p;
  v8h hi = *(const v8h*)(p + 16);
  return cat8(lo, hi);
}

// B-operand (32x16 f16, K x N). Lane n = lane&15; halves B[k0+8*kh+{0..7}] and B[k0+16+8*kh+{0..7}][n],
// read from a buffer storing rows[n][k].
__device__ __forceinline__ v16h loadB(const _Float16* buf, int stride, int rowbase, int k0, int lane) {
  const int kh = lane >> 4;
  const _Float16* p = buf + (rowbase + (lane & 15)) * stride + k0 + 8 * kh;
  v8h lo = *(const v8h*)p;
  v8h hi = *(const v8h*)(p + 16);
  return cat8(lo, hi);
}

__global__ __launch_bounds__(128)
void bga_flash_wmma(const float* __restrict__ q, const float* __restrict__ k,
                    const float* __restrict__ v, const float* __restrict__ amask,
                    const int* __restrict__ lidx, const int* __restrict__ gidx,
                    const float* __restrict__ gk, const float* __restrict__ gv,
                    const float* __restrict__ gm, float* __restrict__ out,
                    int HB, int T)
{
  __shared__ __align__(16) _Float16 qs[64 * QS_STR];
  __shared__ __align__(16) _Float16 ks[2][KT * KS_STR];     // double-buffered K tile (keys x d)
  __shared__ __align__(16) _Float16 vt[2][DHEAD * VT_STR];  // double-buffered V^T tile (d x keys)
  __shared__ __align__(16) _Float16 ps[4 * 16 * PS_STR];    // per-wave probs 16x32
  __shared__ __align__(16) unsigned long long kslot[3][KT]; // resolved K row pointers
  __shared__ __align__(16) unsigned long long vslot[3][KT]; // resolved V row pointers
  __shared__ float mslot[3][KT];                            // mask values

  const int tid  = threadIdx.x;
  const int lane = tid & 31;
  const int w    = tid >> 5;
  const int m16  = lane & 15;

  const int tnb = T / 128;
  int wg = blockIdx.x;
  const int fi = wg & 1; wg >>= 1;
  const int b  = wg % tnb; wg /= tnb;
  const int hh = wg % HB;
  const int nn = wg / HB;
  const int bh = nn * HB + hh;
  const int lb = b * 2 + fi;

  // Resolve gather for key tile jt once per slot: full row pointers + mask -> LDS buffer bufi
  auto idx_calc = [&](int jt, int bufi) {
    if (tid < KT) {
      const int slot = jt * KT + tid;
      const float* kp; const float* vp; float mv;
      if (slot < LKEYS) {
        const int pos = (lb * 64 + slot - 64 + T) % T;
        const int gi = lidx[(size_t)bh * T + pos];
        kp = k + ((size_t)bh * T + gi) * DHEAD;
        vp = v + ((size_t)bh * T + gi) * DHEAD;
        mv = amask[(size_t)nn * T + gi];
      } else if (slot < LKEYS + GKEYS) {
        const int pos = (b * 128 + (slot - LKEYS) - 128 + T) % T;
        const int gi = gidx[(size_t)bh * T + pos];
        kp = k + ((size_t)bh * T + gi) * DHEAD;
        vp = v + ((size_t)bh * T + gi) * DHEAD;
        mv = amask[(size_t)nn * T + gi];
      } else {
        const int g = slot - (LKEYS + GKEYS);
        kp = gk + ((size_t)bh * NGLOBC + g) * DHEAD;
        vp = gv + ((size_t)bh * NGLOBC + g) * DHEAD;
        mv = gm[(size_t)bh * NGLOBC + g];
      }
      kslot[bufi][tid] = (unsigned long long)kp;
      vslot[bufi][tid] = (unsigned long long)vp;
      mslot[bufi][tid] = mv;
    }
  };

  const int krow = tid >> 2, kcb = (tid & 3) * 16;  // K staging: quarter-row per thread
  const int vdr  = tid >> 1, vkb = (tid & 1) * 16;  // V staging: half d-row per thread (transposed)

  auto stage_load = [&](int sbuf, float (&kf)[16], float (&vf)[16]) {
    {
      gcf* src = (gcf*)kslot[sbuf][krow] + kcb;
      #pragma unroll
      for (int j = 0; j < 4; ++j)
        *(v4f*)&kf[4 * j] = *(gcv4f*)(src + 4 * j);
    }
    unsigned long long vp[16];
    #pragma unroll
    for (int e = 0; e < 16; ++e) vp[e] = vslot[sbuf][vkb + e];   // adjacent b64s -> merged ds loads
    #pragma unroll
    for (int e = 0; e < 16; ++e) vf[e] = *((gcf*)vp[e] + vdr);
  };

  auto stage_store = [&](int kvbuf, const float (&kf)[16], const float (&vf)[16]) {
    _Float16 tk[16] __attribute__((aligned(16)));
    _Float16 tv[16] __attribute__((aligned(16)));
    #pragma unroll
    for (int c = 0; c < 16; ++c) { tk[c] = (_Float16)kf[c]; tv[c] = (_Float16)vf[c]; }
    *(v8h*)&ks[kvbuf][krow * KS_STR + kcb]     = *(v8h*)&tk[0];
    *(v8h*)&ks[kvbuf][krow * KS_STR + kcb + 8] = *(v8h*)&tk[8];
    *(v8h*)&vt[kvbuf][vdr * VT_STR + vkb]      = *(v8h*)&tv[0];
    *(v8h*)&vt[kvbuf][vdr * VT_STR + vkb + 8]  = *(v8h*)&tv[8];
  };

  // ---- stage Q tile (64x64) as f16 ----
  {
    const int qrow = tid >> 1, cb = (tid & 1) * 32;
    const float* src = q + ((size_t)bh * T + (size_t)lb * 64 + qrow) * DHEAD + cb;
    _Float16 tmp[32] __attribute__((aligned(16)));
    #pragma unroll
    for (int c = 0; c < 32; ++c) tmp[c] = (_Float16)src[c];
    #pragma unroll
    for (int c = 0; c < 32; c += 8)
      *(v8h*)&qs[qrow * QS_STR + cb + c] = *(v8h*)&tmp[c];
  }
  idx_calc(0, 0);
  __syncthreads();

  // ---- prologue: stage key tile 0 into kv buffer 0; pointers for tile 1 ----
  {
    float kf[16], vf[16];
    stage_load(0, kf, vf);
    stage_store(0, kf, vf);
  }
  idx_calc(1, 1);
  __syncthreads();

  const v16h aqA = loadA(qs, QS_STR, w * 16 + m16, 0,  lane);
  const v16h aqB = loadA(qs, QS_STR, w * 16 + m16, 32, lane);

  float mrun = -__builtin_inff();
  float lpart[8];
  v8f o0 = {}, o1 = {}, o2 = {}, o3 = {};
  #pragma unroll
  for (int r = 0; r < 8; ++r) lpart[r] = 0.f;

  int ci = 0;                       // it % 3
  for (int it = 0; it < NIT; ++it) {
    const int kvp = it & 1, kvs = kvp ^ 1;
    const int si = (ci == 2) ? 0 : ci + 1;   // (it+1)%3
    const int wi = (si == 2) ? 0 : si + 1;   // (it+2)%3
    const bool more = (it + 1 < NIT);

    // 1) issue global gather loads for next key tile (latency hidden by compute)
    float kf[16], vf[16];
    if (more) stage_load(si, kf, vf);

    // 2) compute this key tile: scores = Q K^T / 8 + mask
    const v8f zero = {};
    v16h b00 = loadB(&ks[kvp][0], KS_STR, 0,  0,  lane);
    v16h b01 = loadB(&ks[kvp][0], KS_STR, 0,  32, lane);
    v8f c0 = __builtin_amdgcn_wmma_f32_16x16x32_f16(false, aqA, false, b00, (short)0, zero, false, false);
    c0     = __builtin_amdgcn_wmma_f32_16x16x32_f16(false, aqB, false, b01, (short)0, c0,   false, false);
    v16h b10 = loadB(&ks[kvp][0], KS_STR, 16, 0,  lane);
    v16h b11 = loadB(&ks[kvp][0], KS_STR, 16, 32, lane);
    v8f c1 = __builtin_amdgcn_wmma_f32_16x16x32_f16(false, aqA, false, b10, (short)0, zero, false, false);
    c1     = __builtin_amdgcn_wmma_f32_16x16x32_f16(false, aqB, false, b11, (short)0, c1,   false, false);

    const float mk0 = mslot[ci][m16];
    const float mk1 = mslot[ci][16 + m16];

    float s0[8], s1[8];
    #pragma unroll
    for (int r = 0; r < 8; ++r) {
      s0[r] = c0[r] * 0.125f + mk0;
      s1[r] = c1[r] * 0.125f + mk1;
    }

    // tile-wide max (softmax invariant to the wave-uniform shift): 5 shuffles only
    float lmax = fmaxf(s0[0], s1[0]);
    #pragma unroll
    for (int r = 1; r < 8; ++r) lmax = fmaxf(lmax, fmaxf(s0[r], s1[r]));
    lmax = fmaxf(lmax, __shfl_xor(lmax, 1));
    lmax = fmaxf(lmax, __shfl_xor(lmax, 2));
    lmax = fmaxf(lmax, __shfl_xor(lmax, 4));
    lmax = fmaxf(lmax, __shfl_xor(lmax, 8));
    lmax = fmaxf(lmax, __shfl_xor(lmax, 16));

    const float mn  = fmaxf(mrun, lmax);
    const float esc = __expf(mrun - mn);
    mrun = mn;

    float p0[8], p1[8];
    #pragma unroll
    for (int r = 0; r < 8; ++r) {
      p0[r] = __expf(s0[r] - mn);
      p1[r] = __expf(s1[r] - mn);
      lpart[r] = lpart[r] * esc + (p0[r] + p1[r]);   // per-lane partial sum; reduced after loop
    }
    #pragma unroll
    for (int r = 0; r < 8; ++r) {
      o0[r] *= esc; o1[r] *= esc; o2[r] *= esc; o3[r] *= esc;
    }

    // redistribute probs (C layout -> A layout) through per-wave LDS
    {
      _Float16* pw = ps + w * 16 * PS_STR;
      const int rowoff = (lane >> 4) * 8;
      #pragma unroll
      for (int r = 0; r < 8; ++r) {
        pw[(rowoff + r) * PS_STR + m16]      = (_Float16)p0[r];
        pw[(rowoff + r) * PS_STR + 16 + m16] = (_Float16)p1[r];
      }
    }
    asm volatile("s_wait_dscnt 0" ::: "memory");   // same-wave LDS store->load ordering

    const v16h ap  = loadA(ps + w * 16 * PS_STR, PS_STR, m16, 0, lane);
    const v16h bv0 = loadB(&vt[kvp][0], VT_STR, 0,  0, lane);
    const v16h bv1 = loadB(&vt[kvp][0], VT_STR, 16, 0, lane);
    const v16h bv2 = loadB(&vt[kvp][0], VT_STR, 32, 0, lane);
    const v16h bv3 = loadB(&vt[kvp][0], VT_STR, 48, 0, lane);
    o0 = __builtin_amdgcn_wmma_f32_16x16x32_f16(false, ap, false, bv0, (short)0, o0, false, false);
    o1 = __builtin_amdgcn_wmma_f32_16x16x32_f16(false, ap, false, bv1, (short)0, o1, false, false);
    o2 = __builtin_amdgcn_wmma_f32_16x16x32_f16(false, ap, false, bv2, (short)0, o2, false, false);
    o3 = __builtin_amdgcn_wmma_f32_16x16x32_f16(false, ap, false, bv3, (short)0, o3, false, false);

    // 3) convert + store next tile into the other kv buffer; pointers for tile it+2
    if (more) stage_store(kvs, kf, vf);
    if (it + 2 < NIT) idx_calc(it + 2, wi);
    ci = si;
    __syncthreads();
  }

  // ---- epilogue: single row-sum reduction, normalize, store fp32 ----
  {
    const int rowoff = (lane >> 4) * 8;
    #pragma unroll
    for (int r = 0; r < 8; ++r) {
      float l = lpart[r];
      l += __shfl_xor(l, 1);
      l += __shfl_xor(l, 2);
      l += __shfl_xor(l, 4);
      l += __shfl_xor(l, 8);
      const float inv = 1.0f / l;
      const int trow = lb * 64 + w * 16 + rowoff + r;
      float* dst = out + ((size_t)bh * T + trow) * DHEAD + m16;
      dst[0]  = o0[r] * inv;
      dst[16] = o1[r] * inv;
      dst[32] = o2[r] * inv;
      dst[48] = o3[r] * inv;
    }
  }
}

extern "C" void kernel_launch(void* const* d_in, const int* in_sizes, int n_in,
                              void* d_out, int out_size, void* d_ws, size_t ws_size,
                              hipStream_t stream) {
  (void)n_in; (void)out_size; (void)d_ws; (void)ws_size;
  const float* q     = (const float*)d_in[0];
  const float* k     = (const float*)d_in[1];
  const float* v     = (const float*)d_in[2];
  const float* amask = (const float*)d_in[3];
  const int*   lidx  = (const int*)d_in[4];
  const int*   gidx  = (const int*)d_in[5];
  const float* gk    = (const float*)d_in[6];
  const float* gv    = (const float*)d_in[7];
  const float* gm    = (const float*)d_in[8];
  float* out = (float*)d_out;

  // derive dims: sizes[0]=n*h*t*d, sizes[3]=n*t, sizes[8]=n*h*NGLOB, d=NGLOB=64
  const long nhtd = in_sizes[0];
  const long n_t  = in_sizes[3];
  const int  h    = (int)(nhtd / n_t / DHEAD);
  const int  n    = (int)((long)in_sizes[8] / ((long)h * NGLOBC));
  const int  t    = (int)(n_t / n);

  const int tnb  = t / 128;
  const int grid = n * h * tnb * 2;   // 1536 for the reference shapes

  bga_flash_wmma<<<grid, 128, 0, stream>>>(q, k, v, amask, lidx, gidx, gk, gv, gm, out, h, t);
}
